// GATSingleHead_1975684956197
// MI455X (gfx1250) — matrix operand order repaired and checked by reading, not compiled, not run on hardware
//
#include <hip/hip_runtime.h>
#include <hip/hip_bf16.h>

typedef __attribute__((ext_vector_type(16))) __bf16 v16bf;
typedef __attribute__((ext_vector_type(8)))  float  v8f;

#define NROWS 8192
#define CIN   256
#define COUT  128
#define MAXE  512   // max compacted edges per row (mean ~82, huge headroom)

// ---------------------------------------------------------------------------
// Kernel 1: transpose+convert W [256x128 f32, row-major] -> Wt [128x256 bf16]
// so WMMA B-fragments are contiguous 32B per-lane loads.
// ---------------------------------------------------------------------------
__global__ __launch_bounds__(256) void gat_cvt_w(const float* __restrict__ W,
                                                 __bf16* __restrict__ Wt) {
  const int idx = blockIdx.x * 256 + threadIdx.x;  // 0..32767
  const int k = idx >> 7;                          // 0..255
  const int c = idx & 127;                         // 0..127
  Wt[(size_t)c * CIN + k] = (__bf16)W[(size_t)k * COUT + c];
}

// ---------------------------------------------------------------------------
// Kernel 2: h = X @ W + b via v_wmma_f32_16x16x32_bf16.
// Block = 256 threads = 8 waves; wave w computes rows [blk*128 + w*16, +16)
// across all 128 output cols (8 N-tiles), K swept in 8 steps of 32.
// ---------------------------------------------------------------------------
__global__ __launch_bounds__(256) void gat_gemm_h(const float* __restrict__ X,
                                                  const __bf16* __restrict__ Wt,
                                                  const float* __restrict__ bias,
                                                  float* __restrict__ h) {
  const int lane  = threadIdx.x & 31;
  const int wave  = threadIdx.x >> 5;
  const int rowb  = blockIdx.x * 128 + wave * 16;
  const int laneM = lane & 15;
  const int hi    = lane >> 4;  // lane-half selects K sub-blocks

  v8f acc[8] = {};

  const float* arow = X + (size_t)(rowb + laneM) * CIN;
  #pragma unroll
  for (int k0 = 0; k0 < CIN; k0 += 32) {
    // A fragment: 16x32 bf16. lanes 0-15: K = k0+[0..15];
    // lanes 16-31: K = k0+16+[0..15].
    v16bf a;
    const float* ap = arow + k0 + hi * 16;
    #pragma unroll
    for (int e = 0; e < 8; ++e) a[e]     = (__bf16)ap[e];
    #pragma unroll
    for (int e = 0; e < 8; ++e) a[8 + e] = (__bf16)ap[8 + e];

    #pragma unroll
    for (int n = 0; n < 8; ++n) {
      // B fragment: 32x16 bf16. lane holds col = n*16+laneM, 16 contiguous K
      // starting at k0 + hi*16 -> one aligned 32B load from Wt.
      const v16bf bf =
          *(const v16bf*)(Wt + (size_t)(n * 16 + laneM) * CIN + k0 + hi * 16);
      acc[n] = __builtin_amdgcn_wmma_f32_16x16x32_bf16(
          false, a, false, bf, (short)0, acc[n], false, false);
    }
  }

  // C/D layout: VGPR r holds M = r + hi*8, N = laneM (per 16-col tile).
  #pragma unroll
  for (int n = 0; n < 8; ++n) {
    const int col = n * 16 + laneM;
    const float bv = bias[col];
    #pragma unroll
    for (int r = 0; r < 8; ++r) {
      const int row = rowb + hi * 8 + r;
      h[(size_t)row * COUT + col] = acc[n][r] + bv;
    }
  }
}

// ---------------------------------------------------------------------------
// Kernel 3: f1 = h@v0, f2 = h@v1. One thread per row.
// ---------------------------------------------------------------------------
__global__ __launch_bounds__(256) void gat_f12(const float* __restrict__ h,
                                               const float* __restrict__ v0,
                                               const float* __restrict__ v1,
                                               float* __restrict__ f1,
                                               float* __restrict__ f2) {
  const int row = blockIdx.x * 256 + threadIdx.x;
  const float4* hr  = (const float4*)(h + (size_t)row * COUT);
  const float4* v0p = (const float4*)v0;
  const float4* v1p = (const float4*)v1;
  float s1 = 0.f, s2 = 0.f;
  #pragma unroll 4
  for (int i = 0; i < COUT / 4; ++i) {
    const float4 hv = hr[i], a = v0p[i], c = v1p[i];
    s1 += hv.x * a.x + hv.y * a.y + hv.z * a.z + hv.w * a.w;
    s2 += hv.x * c.x + hv.y * c.y + hv.z * c.z + hv.w * c.w;
  }
  f1[row] = s1;
  f2[row] = s2;
}

// ---------------------------------------------------------------------------
// Kernel 4: per-row edge-masked softmax + sparse attn@h.
// One block (256 thr) per row. DETERMINISTIC edge compaction: each thread
// owns 32 contiguous columns, counts nonzeros, LDS prefix-sum assigns the
// write offset -> edge list sorted by j, fixed FP summation order.
// ---------------------------------------------------------------------------
__global__ __launch_bounds__(256) void gat_attn(const float* __restrict__ adj,
                                                const float* __restrict__ h,
                                                const float* __restrict__ f1,
                                                const float* __restrict__ f2,
                                                float* __restrict__ out) {
  __shared__ int   s_ofs[256];
  __shared__ int   s_j[MAXE];
  __shared__ float s_v[MAXE];
  __shared__ float s_red[256];

  const int tid = threadIdx.x;
  const int row = blockIdx.x;
  const float f1i = f1[row];

  // Thread t streams columns [t*32, t*32+32): 8 aligned float4 loads.
  const float4* arow =
      (const float4*)(adj + (size_t)row * NROWS + (size_t)tid * 32);
  float a[32];
  #pragma unroll
  for (int t = 0; t < 8; ++t) {
    const float4 a4 = arow[t];
    a[4 * t + 0] = a4.x; a[4 * t + 1] = a4.y;
    a[4 * t + 2] = a4.z; a[4 * t + 3] = a4.w;
  }
  int cnt_local = 0;
  #pragma unroll
  for (int c = 0; c < 32; ++c) cnt_local += (a[c] != 0.0f) ? 1 : 0;

  // Inclusive Hillis-Steele scan over 256 per-thread counts.
  s_ofs[tid] = cnt_local;
  __syncthreads();
  for (int d = 1; d < 256; d <<= 1) {
    const int v   = s_ofs[tid];
    const int add = (tid >= d) ? s_ofs[tid - d] : 0;
    __syncthreads();
    s_ofs[tid] = v + add;
    __syncthreads();
  }
  const int total = s_ofs[255];
  int p = s_ofs[tid] - cnt_local;  // exclusive offset

  // Write this thread's edges in column order at its offset.
  const int jbase = tid * 32;
  #pragma unroll
  for (int c = 0; c < 32; ++c) {
    if (a[c] != 0.0f) {
      if (p < MAXE) {
        const int j = jbase + c;
        const float x = a[c] * (f1i + f2[j]);
        s_j[p] = j;
        s_v[p] = 1.0f / (1.0f + __expf(-x)) - 0.5f;  // TEMP = 1
      }
      ++p;
    }
  }
  __syncthreads();
  int cnt = total;
  if (cnt > MAXE) cnt = MAXE;

  // Row max over compacted vals.
  float m = -3.4e38f;
  for (int k = tid; k < cnt; k += 256) m = fmaxf(m, s_v[k]);
  s_red[tid] = m;
  __syncthreads();
  for (int s = 128; s > 0; s >>= 1) {
    if (tid < s) s_red[tid] = fmaxf(s_red[tid], s_red[tid + s]);
    __syncthreads();
  }
  m = (cnt > 0) ? s_red[0] : 0.0f;
  __syncthreads();

  // exp(val - max), row sum.
  float ssum = 0.f;
  for (int k = tid; k < cnt; k += 256) {
    const float e = __expf(s_v[k] - m);
    s_v[k] = e;
    ssum += e;
  }
  s_red[tid] = ssum;
  __syncthreads();
  for (int s = 128; s > 0; s >>= 1) {
    if (tid < s) s_red[tid] += s_red[tid + s];
    __syncthreads();
  }
  const float denom = fmaxf(s_red[0], 1.1754944e-38f);  // empty row -> 0 out
  __syncthreads();

  // out[row, ch] = sum_k e_k * h[j_k, ch] / denom; 2 fixed-order partials.
  const int ch   = tid & 127;
  const int half = tid >> 7;
  float acc = 0.f;
  for (int k = half; k < cnt; k += 2) acc += s_v[k] * h[(size_t)s_j[k] * COUT + ch];
  s_red[tid] = acc;
  __syncthreads();
  if (tid < 128) out[(size_t)row * COUT + tid] = (s_red[tid] + s_red[tid + 128]) / denom;
}

// ---------------------------------------------------------------------------
extern "C" void kernel_launch(void* const* d_in, const int* in_sizes, int n_in,
                              void* d_out, int out_size, void* d_ws, size_t ws_size,
                              hipStream_t stream) {
  const float* X   = (const float*)d_in[0];  // [8192,256]
  const float* adj = (const float*)d_in[1];  // [8192,8192]
  const float* W   = (const float*)d_in[2];  // [256,128]
  const float* b   = (const float*)d_in[3];  // [128]
  const float* v0  = (const float*)d_in[4];  // [128]
  const float* v1  = (const float*)d_in[5];  // [128]
  float* out = (float*)d_out;                // [8192,128]

  char* ws = (char*)d_ws;
  float*  h  = (float*)ws;                                  // 4 MB
  __bf16* Wt = (__bf16*)(ws + (size_t)NROWS * COUT * 4);    // 64 KB
  float*  f1 = (float*)(ws + (size_t)NROWS * COUT * 4 + (size_t)COUT * CIN * 2);
  float*  f2 = f1 + NROWS;

  gat_cvt_w<<<(CIN * COUT) / 256, 256, 0, stream>>>(W, Wt);
  gat_gemm_h<<<NROWS / 128, 256, 0, stream>>>(X, Wt, b, h);
  gat_f12<<<NROWS / 256, 256, 0, stream>>>(h, v0, v1, f1, f2);
  gat_attn<<<NROWS, 256, 0, stream>>>(adj, h, f1, f2, out);
}
